// VilT_Classification_32246614459160
// MI455X (gfx1250) — matrix operand order repaired and checked by reading, not compiled, not run on hardware
//
#include <hip/hip_runtime.h>
#include <hip/hip_bf16.h>
#include <math.h>

// ---------------------------------------------------------------------------
// Types for CDNA5 WMMA (wave32, v_wmma_f32_16x16x32_bf16)
// ---------------------------------------------------------------------------
typedef __bf16 bf16_t;
typedef bf16_t v16bf __attribute__((ext_vector_type(16)));
typedef bf16_t v8bf  __attribute__((ext_vector_type(8)));
typedef float  v8f   __attribute__((ext_vector_type(8)));

#define H768 768
#define G2304 2304

// ---------------------------------------------------------------------------
// WMMA GEMM:  C[M,N] = A[M,K](bf16) * W[N,K](bf16)^T + bias
// Block tile 128x128, 512 threads = 16 waves in a 4x4 grid, each wave owns a
// 2x2 grid of 16x16 accumulators (fragment reuse 2x -> ~1KB LDS per wmma).
// W is [N,K] row-major, which is exactly the K-contiguous layout the
// B-fragment wants (ISA 7.12.2), so all LDS fragment reads are b128s.
// ---------------------------------------------------------------------------
__global__ __launch_bounds__(512)
void wmma_gemm_bf16(const bf16_t* __restrict__ A, int lda,
                    const bf16_t* __restrict__ W, int ldw,
                    float* __restrict__ Cf, bf16_t* __restrict__ Cb,
                    const float* __restrict__ bias,
                    int M, int N, int K) {
  __shared__ bf16_t As[128 * 64];
  __shared__ bf16_t Ws[128 * 64];

  const int tid  = threadIdx.x;
  const int lane = tid & 31;
  const int wave = tid >> 5;          // 0..15
  const int wm   = wave >> 2;         // 0..3
  const int wn   = wave & 3;          // 0..3
  const int half = lane >> 4;         // 0/1
  const int lr   = lane & 15;

  const int m0 = blockIdx.y * 128;
  const int n0 = blockIdx.x * 128;

  // global->LDS: each thread moves 16 bf16 (2x 16B) per tile
  const int ldr = tid >> 2;           // 0..127
  const int ldc = (tid & 3) * 16;     // 0,16,32,48

  const v8f vzero = {0.f, 0.f, 0.f, 0.f, 0.f, 0.f, 0.f, 0.f};
  v8f acc[2][2];
  acc[0][0] = vzero; acc[0][1] = vzero; acc[1][0] = vzero; acc[1][1] = vzero;

  for (int k0 = 0; k0 < K; k0 += 64) {
    {
      int gr = m0 + ldr;
      uint4 v0 = make_uint4(0u, 0u, 0u, 0u), v1 = v0;
      if (gr < M) {
        const uint4* p = (const uint4*)(A + (size_t)gr * lda + k0 + ldc);
        v0 = p[0]; v1 = p[1];
      }
      uint4* q = (uint4*)(As + ldr * 64 + ldc);
      q[0] = v0; q[1] = v1;
    }
    {
      int gr = n0 + ldr;
      uint4 v0 = make_uint4(0u, 0u, 0u, 0u), v1 = v0;
      if (gr < N) {
        const uint4* p = (const uint4*)(W + (size_t)gr * ldw + k0 + ldc);
        v0 = p[0]; v1 = p[1];
      }
      uint4* q = (uint4*)(Ws + ldr * 64 + ldc);
      q[0] = v0; q[1] = v1;
    }
    __syncthreads();

#pragma unroll
    for (int kk = 0; kk < 64; kk += 32) {
      v16bf afrag[2], bfrag[2];
#pragma unroll
      for (int i2 = 0; i2 < 2; ++i2) {
        // A fragment: lane l holds row M=l%16, K = half*8+0..7 and 16+half*8+0..7
        const v8bf a0 = *(const v8bf*)(As + ((wm * 2 + i2) * 16 + lr) * 64 + kk + half * 8);
        const v8bf a1 = *(const v8bf*)(As + ((wm * 2 + i2) * 16 + lr) * 64 + kk + 16 + half * 8);
        // B fragment: lane l holds col N=l%16, K = half*8+0..7 and 16+half*8+0..7
        const v8bf b0 = *(const v8bf*)(Ws + ((wn * 2 + i2) * 16 + lr) * 64 + kk + half * 8);
        const v8bf b1 = *(const v8bf*)(Ws + ((wn * 2 + i2) * 16 + lr) * 64 + kk + 16 + half * 8);
#pragma unroll
        for (int i = 0; i < 8; ++i) {
          afrag[i2][i] = a0[i]; afrag[i2][8 + i] = a1[i];
          bfrag[i2][i] = b0[i]; bfrag[i2][8 + i] = b1[i];
        }
      }
#pragma unroll
      for (int mt = 0; mt < 2; ++mt)
#pragma unroll
        for (int nt = 0; nt < 2; ++nt)
          acc[mt][nt] = __builtin_amdgcn_wmma_f32_16x16x32_bf16(
              false, afrag[mt], false, bfrag[nt], (short)0, acc[mt][nt],
              false, false);
    }
    __syncthreads();
  }

  // C/D layout: lane l -> N = l%16, VGPR v -> M = (l>=16 ? 8 : 0) + v
#pragma unroll
  for (int mt = 0; mt < 2; ++mt) {
#pragma unroll
    for (int nt = 0; nt < 2; ++nt) {
      int n = n0 + (wn * 2 + nt) * 16 + lr;
      float bv = (bias != nullptr && n < N) ? bias[n] : 0.f;
      int mbase = m0 + (wm * 2 + mt) * 16 + half * 8;
#pragma unroll
      for (int v = 0; v < 8; ++v) {
        int m = mbase + v;
        if (m < M && n < N) {
          float val = acc[mt][nt][v] + bv;
          if (Cf) Cf[(size_t)m * N + n] = val;
          if (Cb) Cb[(size_t)m * N + n] = (bf16_t)val;
        }
      }
    }
  }
}

// ---------------------------------------------------------------------------
// Fused GRU recurrent step (single launch per timestep):
//   gh(r,z,n) = h_{t-1} @ Whh^T  done with WMMA, three gate W-tiles staged in
//   LDS (A fragment reused 3x), gate nonlinearity in the epilogue.
// Grid (H/64, B/64) = (12,4), 512 threads = 16 waves (4x4 of 16x16 tiles).
// H state is ping-pong buffered across t (blocks read h_{t-1}, write h_t).
// GI already contains bih; acc += wvec[t]*h_t folds text_embed / feats.
// ---------------------------------------------------------------------------
__global__ __launch_bounds__(512)
void gru_step_fused(const bf16_t* __restrict__ HbIn, const float* __restrict__ Hin,
                    const bf16_t* __restrict__ Whh, const float* __restrict__ bhh,
                    const bf16_t* __restrict__ GI,
                    float* __restrict__ Hout, bf16_t* __restrict__ HbOut,
                    float* __restrict__ acc, const float* __restrict__ wvec,
                    int T, int t) {
  __shared__ bf16_t As[64 * 64];
  __shared__ bf16_t Ws[3 * 64 * 64];

  const int tid  = threadIdx.x;
  const int lane = tid & 31;
  const int wave = tid >> 5;
  const int wm   = wave >> 2;
  const int wn   = wave & 3;
  const int half = lane >> 4;
  const int lr   = lane & 15;

  const int m0 = blockIdx.y * 64;     // batch rows (0..192)
  const int n0 = blockIdx.x * 64;     // hidden cols (0..704)

  const int ldr = tid >> 3;           // 0..63
  const int ldc = (tid & 7) * 8;      // 0..56

  const v8f vzero = {0.f, 0.f, 0.f, 0.f, 0.f, 0.f, 0.f, 0.f};
  v8f gacc[3];
  gacc[0] = vzero; gacc[1] = vzero; gacc[2] = vzero;

  for (int k0 = 0; k0 < H768; k0 += 64) {
    *(uint4*)(As + ldr * 64 + ldc) =
        *(const uint4*)(HbIn + (size_t)(m0 + ldr) * H768 + k0 + ldc);
#pragma unroll
    for (int g = 0; g < 3; ++g)
      *(uint4*)(Ws + g * 4096 + ldr * 64 + ldc) =
          *(const uint4*)(Whh + (size_t)(g * H768 + n0 + ldr) * H768 + k0 + ldc);
    __syncthreads();

#pragma unroll
    for (int kk = 0; kk < 64; kk += 32) {
      const v8bf a0 = *(const v8bf*)(As + (wm * 16 + lr) * 64 + kk + half * 8);
      const v8bf a1 = *(const v8bf*)(As + (wm * 16 + lr) * 64 + kk + 16 + half * 8);
      v16bf afr;
#pragma unroll
      for (int i = 0; i < 8; ++i) { afr[i] = a0[i]; afr[8 + i] = a1[i]; }
#pragma unroll
      for (int g = 0; g < 3; ++g) {
        const v8bf b0 = *(const v8bf*)(Ws + g * 4096 + (wn * 16 + lr) * 64 + kk + half * 8);
        const v8bf b1 = *(const v8bf*)(Ws + g * 4096 + (wn * 16 + lr) * 64 + kk + 16 + half * 8);
        v16bf bfr;
#pragma unroll
        for (int i = 0; i < 8; ++i) { bfr[i] = b0[i]; bfr[8 + i] = b1[i]; }
        gacc[g] = __builtin_amdgcn_wmma_f32_16x16x32_bf16(
            false, afr, false, bfr, (short)0, gacc[g], false, false);
      }
    }
    __syncthreads();
  }

  // gate epilogue
  const int j = n0 + wn * 16 + lr;
  const float bhr = bhh[j];
  const float bhz = bhh[H768 + j];
  const float bhn = bhh[2 * H768 + j];
  const float wt  = wvec[t];
  const int mbase = m0 + wm * 16 + half * 8;
#pragma unroll
  for (int v = 0; v < 8; ++v) {
    int b = mbase + v;
    size_t gi = ((size_t)b * T + t) * G2304;
    float ir  = (float)GI[gi + j];
    float iz  = (float)GI[gi + H768 + j];
    float inn = (float)GI[gi + 2 * H768 + j];
    float r = 1.f / (1.f + __expf(-(ir + gacc[0][v] + bhr)));
    float z = 1.f / (1.f + __expf(-(iz + gacc[1][v] + bhz)));
    float nn = tanhf(inn + r * (gacc[2][v] + bhn));
    size_t hi = (size_t)b * H768 + j;
    float hnew = (1.f - z) * nn + z * Hin[hi];
    Hout[hi] = hnew;
    HbOut[hi] = (bf16_t)hnew;
    acc[hi] += wt * hnew;
  }
}

// ---------------------------------------------------------------------------
// Packing / casting kernels
// ---------------------------------------------------------------------------
__global__ void cast_f32_bf16(const float* __restrict__ src,
                              bf16_t* __restrict__ dst, int n) {
  int i = blockIdx.x * blockDim.x + threadIdx.x;
  if (i < n) dst[i] = (bf16_t)src[i];
}

__global__ void pack_s3d(const float* __restrict__ t0,
                         const float* __restrict__ t1,
                         bf16_t* __restrict__ out) {
  int row = blockIdx.x;   // 0..7679, b = row/30
  const float* src = (row < 128 * 30) ? (t0 + (size_t)row * 1024)
                                      : (t1 + (size_t)(row - 128 * 30) * 1024);
  bf16_t* dst = out + (size_t)row * 1024;
  for (int j = threadIdx.x; j < 1024; j += blockDim.x) dst[j] = (bf16_t)src[j];
}

__global__ void pack_text(const float* __restrict__ t0,
                          const float* __restrict__ t1,
                          bf16_t* __restrict__ out) {
  int row = blockIdx.x;   // 0..18687, b = row/73
  const float* src = (row < 128 * 73) ? (t0 + (size_t)row * H768)
                                      : (t1 + (size_t)(row - 128 * 73) * H768);
  bf16_t* dst = out + (size_t)row * H768;
  for (int j = threadIdx.x; j < H768; j += blockDim.x) dst[j] = (bf16_t)src[j];
}

// GRU init: H = 0, Hb = 0, acc = scalar bias (b_dt / b_c1); 768 blocks x 256
__global__ void gru_init(float* __restrict__ Hst, bf16_t* __restrict__ Hb,
                         float* __restrict__ acc,
                         const float* __restrict__ bias_scalar) {
  int i = blockIdx.x * blockDim.x + threadIdx.x;
  Hst[i] = 0.f;
  Hb[i] = (bf16_t)0.f;
  acc[i] = bias_scalar[0];
}

// cV[b][j] = mean_s raw_img[b][s][j]  (equals meanA/meanB on its halves)
__global__ __launch_bounds__(768)
void mean_rows(const float* __restrict__ raw, float* __restrict__ cV) {
  int b = blockIdx.x, j = threadIdx.x;
  float s = 0.f;
  for (int t = 0; t < 30; ++t) s += raw[((size_t)b * 30 + t) * H768 + j];
  cV[(size_t)b * H768 + j] = s * (1.f / 30.f);
}

// d3dot[b] = dot(cV[(b+128)%256], W_d3[768:1536]) + b_d3
__global__ __launch_bounds__(256)
void d3dot_kernel(const float* __restrict__ cV, const float* __restrict__ W_d3,
                  const float* __restrict__ b_d3, float* __restrict__ d3dot) {
  __shared__ float red[256];
  int b = blockIdx.x;
  int partner = (b + 128) & 255;
  float s = 0.f;
  for (int j = threadIdx.x; j < H768; j += 256)
    s += cV[(size_t)partner * H768 + j] * W_d3[H768 + j];
  red[threadIdx.x] = s;
  __syncthreads();
  for (int o = 128; o > 0; o >>= 1) {
    if (threadIdx.x < o) red[threadIdx.x] += red[threadIdx.x + o];
    __syncthreads();
  }
  if (threadIdx.x == 0) d3dot[b] = red[0] + b_d3[0];
}

// per (b,s): l1=<x,text_embed[b]>, l2=<x,cV[b]>, l3=<x,W_d3[0:768]>+d3dot[b]
__global__ __launch_bounds__(256)
void attn_logits(const float* __restrict__ raw, const float* __restrict__ te,
                 const float* __restrict__ cV, const float* __restrict__ W_d3,
                 const float* __restrict__ d3dot,
                 float* __restrict__ l1, float* __restrict__ l2,
                 float* __restrict__ l3) {
  __shared__ float r1[256], r2[256], r3[256];
  int row = blockIdx.x;       // 0..7679
  int b = row / 30;
  const float* x  = raw + (size_t)row * H768;
  const float* tb = te + (size_t)b * H768;
  const float* cb = cV + (size_t)b * H768;
  float s1 = 0.f, s2 = 0.f, s3 = 0.f;
  for (int j = threadIdx.x; j < H768; j += 256) {
    float v = x[j];
    s1 += v * tb[j];
    s2 += v * cb[j];
    s3 += v * W_d3[j];
  }
  r1[threadIdx.x] = s1; r2[threadIdx.x] = s2; r3[threadIdx.x] = s3;
  __syncthreads();
  for (int o = 128; o > 0; o >>= 1) {
    if (threadIdx.x < o) {
      r1[threadIdx.x] += r1[threadIdx.x + o];
      r2[threadIdx.x] += r2[threadIdx.x + o];
      r3[threadIdx.x] += r3[threadIdx.x + o];
    }
    __syncthreads();
  }
  if (threadIdx.x == 0) {
    l1[row] = r1[0];
    l2[row] = r2[0];
    l3[row] = r3[0] + d3dot[b];
  }
}

// softmax over s (30) for l1,l2,l3; comb = p1+p2+p3 ; caus = p3
__global__ __launch_bounds__(32)
void softmax_combine(const float* __restrict__ l1, const float* __restrict__ l2,
                     const float* __restrict__ l3, float* __restrict__ comb,
                     float* __restrict__ caus) {
  __shared__ float sh[32];
  int b = blockIdx.x;
  int s = threadIdx.x;
  const float* Ls[3] = {l1, l2, l3};
  float p[3];
  for (int a = 0; a < 3; ++a) {
    float v = (s < 30) ? Ls[a][b * 30 + s] : -1e30f;
    sh[s] = v; __syncthreads();
    for (int o = 16; o > 0; o >>= 1) {
      if (s < o) sh[s] = fmaxf(sh[s], sh[s + o]);
      __syncthreads();
    }
    float mx = sh[0]; __syncthreads();
    float e = (s < 30) ? __expf(v - mx) : 0.f;
    sh[s] = e; __syncthreads();
    for (int o = 16; o > 0; o >>= 1) {
      if (s < o) sh[s] += sh[s + o];
      __syncthreads();
    }
    float sum = sh[0]; __syncthreads();
    p[a] = e / sum;
  }
  if (s < 30) {
    comb[b * 30 + s] = p[0] + p[1] + p[2];
    caus[b * 30 + s] = p[2];
  }
}

// fusion GRU input (bf16): s<30 -> raw_img*(c*W_d2+b_d2), else all_text
__global__ __launch_bounds__(256)
void build_fusion_x(const float* __restrict__ raw, const float* __restrict__ comb,
                    const float* __restrict__ W_d2, const float* __restrict__ b_d2,
                    const bf16_t* __restrict__ Xtext, bf16_t* __restrict__ Xfus) {
  int row = blockIdx.x;                 // 0..26367
  int b = row / 103, s = row - b * 103;
  bf16_t* dst = Xfus + (size_t)row * H768;
  if (s < 30) {
    float c = comb[b * 30 + s];
    const float* src = raw + ((size_t)b * 30 + s) * H768;
    for (int j = threadIdx.x; j < H768; j += 256)
      dst[j] = (bf16_t)(src[j] * (c * W_d2[j] + b_d2[j]));
  } else {
    const bf16_t* src = Xtext + ((size_t)b * 73 + (s - 30)) * H768;
    for (int j = threadIdx.x; j < H768; j += 256) dst[j] = src[j];
  }
}

// pred[b][o] = b_c2[o] + <leaky(facc[b]), W_c2[o]>   (facc includes b_c1)
__global__ __launch_bounds__(256)
void pred_kernel(const float* __restrict__ facc, const float* __restrict__ W_c2,
                 const float* __restrict__ b_c2, float* __restrict__ out) {
  __shared__ float red[256];
  int idx = blockIdx.x;                // 0..5119
  int b = idx / 20, o = idx - b * 20;
  float s = 0.f;
  for (int j = threadIdx.x; j < H768; j += 256) {
    float f = facc[(size_t)b * H768 + j];
    f = (f >= 0.f) ? f : 0.01f * f;
    s += f * W_c2[(size_t)o * H768 + j];
  }
  red[threadIdx.x] = s;
  __syncthreads();
  for (int r = 128; r > 0; r >>= 1) {
    if (threadIdx.x < r) red[threadIdx.x] += red[threadIdx.x + r];
    __syncthreads();
  }
  if (threadIdx.x == 0) out[idx] = red[0] + b_c2[o];
}

// top-10 indices of caus rows (A half and B half), one thread per b
__global__ void topk_idx(const float* __restrict__ caus, int* __restrict__ idxA,
                         int* __restrict__ idxB) {
  int b = blockIdx.x * blockDim.x + threadIdx.x;
  if (b >= 128) return;
  for (int grp = 0; grp < 2; ++grp) {
    const float* v = caus + (size_t)(grp * 128 + b) * 30;
    float vals[30];
    for (int s = 0; s < 30; ++s) vals[s] = v[s];
    int* dst = (grp ? idxB : idxA) + b * 10;
    for (int k = 0; k < 10; ++k) {
      int best = 0; float bv = -1e30f;
      for (int s = 0; s < 30; ++s)
        if (vals[s] > bv) { bv = vals[s]; best = s; }
      dst[k] = best;
      vals[best] = -1e30f;
    }
  }
}

// Gdiff[b] = mean(top10 A rows) - mean(top10 B rows), cast to bf16
__global__ __launch_bounds__(256)
void gdiff_kernel(const float* __restrict__ raw, const int* __restrict__ idxA,
                  const int* __restrict__ idxB, bf16_t* __restrict__ Gd) {
  int b = blockIdx.x;   // 0..127
  for (int j = threadIdx.x; j < H768; j += 256) {
    float sa = 0.f, sb = 0.f;
    for (int k = 0; k < 10; ++k) {
      sa += raw[((size_t)b * 30 + idxA[b * 10 + k]) * H768 + j];
      sb += raw[((size_t)(128 + b) * 30 + idxB[b * 10 + k]) * H768 + j];
    }
    Gd[(size_t)b * H768 + j] = (bf16_t)((sa - sb) * 0.1f);
  }
}

__global__ void zero_scalar(float* p) { *p = 0.f; }

// lossAB = mean(Dm^2) over 128*768 elements
__global__ __launch_bounds__(256)
void loss_kernel(const float* __restrict__ Dm, float* __restrict__ out) {
  __shared__ float red[256];
  float s = 0.f;
  const int total = 128 * H768;
  for (int i = blockIdx.x * 256 + threadIdx.x; i < total; i += gridDim.x * 256) {
    float v = Dm[i];
    s += v * v;
  }
  red[threadIdx.x] = s;
  __syncthreads();
  for (int o = 128; o > 0; o >>= 1) {
    if (threadIdx.x < o) red[threadIdx.x] += red[threadIdx.x + o];
    __syncthreads();
  }
  if (threadIdx.x == 0) atomicAdd(out, red[0] * (1.f / (128.f * 768.f)));
}

// ---------------------------------------------------------------------------
// Host orchestration
// ---------------------------------------------------------------------------
extern "C" void kernel_launch(void* const* d_in, const int* in_sizes, int n_in,
                              void* d_out, int out_size, void* d_ws, size_t ws_size,
                              hipStream_t stream) {
  const float* s3d      = (const float*)d_in[0];
  const float* textf    = (const float*)d_in[1];
  const float* pos_s3d  = (const float*)d_in[2];
  const float* pos_text = (const float*)d_in[3];
  const float* W_lin1   = (const float*)d_in[4];
  const float* b_lin1   = (const float*)d_in[5];
  const float* Wih_t    = (const float*)d_in[6];
  const float* Whh_t    = (const float*)d_in[7];
  const float* bih_t    = (const float*)d_in[8];
  const float* bhh_t    = (const float*)d_in[9];
  const float* W_dt     = (const float*)d_in[10];
  const float* b_dt     = (const float*)d_in[11];
  const float* W_d3     = (const float*)d_in[12];
  const float* b_d3     = (const float*)d_in[13];
  const float* W_d2     = (const float*)d_in[14];
  const float* b_d2     = (const float*)d_in[15];
  const float* Wih_g    = (const float*)d_in[16];
  const float* Whh_g    = (const float*)d_in[17];
  const float* bih_g    = (const float*)d_in[18];
  const float* bhh_g    = (const float*)d_in[19];
  const float* W_c1     = (const float*)d_in[20];
  const float* b_c1     = (const float*)d_in[21];
  const float* W_c2     = (const float*)d_in[22];
  const float* b_c2     = (const float*)d_in[23];
  const float* W_d4     = (const float*)d_in[24];
  float* out = (float*)d_out;

  // ---- workspace bump allocator (deterministic layout every call) ----
  char* ws = (char*)d_ws;
  size_t off = 0;
  auto alloc = [&](size_t bytes) -> void* {
    void* p = ws + off;
    off += (bytes + 255) & ~(size_t)255;
    return p;
  };

  bf16_t* W_lin1b = (bf16_t*)alloc((size_t)768 * 1024 * 2);
  bf16_t* Wih_tb  = (bf16_t*)alloc((size_t)G2304 * H768 * 2);
  bf16_t* Whh_tb  = (bf16_t*)alloc((size_t)G2304 * H768 * 2);
  bf16_t* Wih_gb  = (bf16_t*)alloc((size_t)G2304 * H768 * 2);
  bf16_t* Whh_gb  = (bf16_t*)alloc((size_t)G2304 * H768 * 2);
  bf16_t* W_d4b   = (bf16_t*)alloc((size_t)H768 * H768 * 2);

  bf16_t* Xs3d    = (bf16_t*)alloc((size_t)7680 * 1024 * 2);
  float*  raw_img = (float*)alloc((size_t)7680 * H768 * 4);
  bf16_t* Xtext   = (bf16_t*)alloc((size_t)18688 * H768 * 2);
  bf16_t* Xfus    = (bf16_t*)alloc((size_t)26368 * H768 * 2);
  bf16_t* GIb     = (bf16_t*)alloc((size_t)26368 * G2304 * 2);  // shared by both GRUs
  float*  H0      = (float*)alloc((size_t)256 * H768 * 4);      // h ping-pong
  float*  H1      = (float*)alloc((size_t)256 * H768 * 4);
  bf16_t* Hb0     = (bf16_t*)alloc((size_t)256 * H768 * 2);
  bf16_t* Hb1     = (bf16_t*)alloc((size_t)256 * H768 * 2);
  float*  te_acc  = (float*)alloc((size_t)256 * H768 * 4);      // text_embed
  float*  facc    = (float*)alloc((size_t)256 * H768 * 4);      // classifier feats
  float*  cV      = (float*)alloc((size_t)256 * H768 * 4);
  float*  l1      = (float*)alloc((size_t)7680 * 4);
  float*  l2      = (float*)alloc((size_t)7680 * 4);
  float*  l3      = (float*)alloc((size_t)7680 * 4);
  float*  d3dot   = (float*)alloc((size_t)256 * 4);
  float*  comb    = (float*)alloc((size_t)7680 * 4);
  float*  caus    = (float*)alloc((size_t)7680 * 4);
  int*    idxA    = (int*)alloc((size_t)128 * 10 * 4);
  int*    idxB    = (int*)alloc((size_t)128 * 10 * 4);
  bf16_t* Gdiff   = (bf16_t*)alloc((size_t)128 * H768 * 2);
  float*  Dm      = (float*)alloc((size_t)128 * H768 * 4);

  auto gemm = [&](const bf16_t* A, int lda, const bf16_t* W, int ldw,
                  float* Cf, bf16_t* Cb, const float* bias,
                  int M, int N, int K) {
    dim3 g((N + 127) / 128, (M + 127) / 128);
    wmma_gemm_bf16<<<g, 512, 0, stream>>>(A, lda, W, ldw, Cf, Cb, bias, M, N, K);
  };

  // ---- cast weights to bf16 ----
  {
    int n;
    n = 768 * 1024;
    cast_f32_bf16<<<(n + 255) / 256, 256, 0, stream>>>(W_lin1, W_lin1b, n);
    n = G2304 * H768;
    cast_f32_bf16<<<(n + 255) / 256, 256, 0, stream>>>(Wih_t, Wih_tb, n);
    cast_f32_bf16<<<(n + 255) / 256, 256, 0, stream>>>(Whh_t, Whh_tb, n);
    cast_f32_bf16<<<(n + 255) / 256, 256, 0, stream>>>(Wih_g, Wih_gb, n);
    cast_f32_bf16<<<(n + 255) / 256, 256, 0, stream>>>(Whh_g, Whh_gb, n);
    n = H768 * H768;
    cast_f32_bf16<<<(n + 255) / 256, 256, 0, stream>>>(W_d4, W_d4b, n);
  }

  // ---- module0: raw_img = [s3d; pos_s3d] @ W_lin1^T + b_lin1 ----
  pack_s3d<<<7680, 256, 0, stream>>>(s3d, pos_s3d, Xs3d);
  gemm(Xs3d, 1024, W_lin1b, 1024, raw_img, nullptr, b_lin1, 7680, H768, 1024);

  // ---- text GRU (T=73): GI hoisted into one big parallel GEMM ----
  pack_text<<<18688, 256, 0, stream>>>(textf, pos_text, Xtext);
  gemm(Xtext, H768, Wih_tb, H768, nullptr, GIb, bih_t, 18688, G2304, H768);
  gru_init<<<768, 256, 0, stream>>>(H0, Hb0, te_acc, b_dt);
  for (int t = 0; t < 73; ++t) {
    const float*  Hin   = (t & 1) ? H1 : H0;
    float*        Hout  = (t & 1) ? H0 : H1;
    const bf16_t* HbIn  = (t & 1) ? Hb1 : Hb0;
    bf16_t*       HbOut = (t & 1) ? Hb0 : Hb1;
    gru_step_fused<<<dim3(12, 4), 512, 0, stream>>>(
        HbIn, Hin, Whh_tb, bhh_t, GIb, Hout, HbOut, te_acc, W_dt, 73, t);
  }

  // ---- attention glue ----
  mean_rows<<<256, 768, 0, stream>>>(raw_img, cV);
  d3dot_kernel<<<256, 256, 0, stream>>>(cV, W_d3, b_d3, d3dot);
  attn_logits<<<7680, 256, 0, stream>>>(raw_img, te_acc, cV, W_d3, d3dot,
                                        l1, l2, l3);
  softmax_combine<<<256, 32, 0, stream>>>(l1, l2, l3, comb, caus);
  build_fusion_x<<<26368, 256, 0, stream>>>(raw_img, comb, W_d2, b_d2,
                                            Xtext, Xfus);

  // ---- fusion GRU (T=103) ----
  gemm(Xfus, H768, Wih_gb, H768, nullptr, GIb, bih_g, 26368, G2304, H768);
  gru_init<<<768, 256, 0, stream>>>(H0, Hb0, facc, b_c1);
  for (int t = 0; t < 103; ++t) {
    const float*  Hin   = (t & 1) ? H1 : H0;
    float*        Hout  = (t & 1) ? H0 : H1;
    const bf16_t* HbIn  = (t & 1) ? Hb1 : Hb0;
    bf16_t*       HbOut = (t & 1) ? Hb0 : Hb1;
    gru_step_fused<<<dim3(12, 4), 512, 0, stream>>>(
        HbIn, Hin, Whh_gb, bhh_g, GIb, Hout, HbOut, facc, W_c1, 103, t);
  }

  // ---- classifier head ----
  pred_kernel<<<5120, 256, 0, stream>>>(facc, W_c2, b_c2, out);

  // ---- top-k loss: (meanTopA - meanTopB) @ W_d4^T, then mean of squares ----
  topk_idx<<<1, 128, 0, stream>>>(caus, idxA, idxB);
  gdiff_kernel<<<128, 256, 0, stream>>>(raw_img, idxA, idxB, Gdiff);
  gemm(Gdiff, H768, W_d4b, H768, Dm, nullptr, nullptr, 128, H768, H768);
  zero_scalar<<<1, 1, 0, stream>>>(out + 5120);
  loss_kernel<<<96, 256, 0, stream>>>(Dm, out + 5120);
}
